// VectorQuantizer_52707838656974
// MI455X (gfx1250) — matrix operand order repaired and checked by reading, not compiled, not run on hardware
//
#include <hip/hip_runtime.h>

// VQ-VAE vector quantizer for MI455X (gfx1250), wave32 + WMMA bf16.
//   latents: [32, 256, 32, 32] f32   (per batch this is x^T: [256 x 1024] contiguous)
//   codebook: [1024, 256] f32
// Outputs (concat, f32): quantized [8388608], vq_loss [1], indices [32768] (as float).
//
// Distance GEMM: split-precision 3x bf16 WMMA (a_h*b_h + a_l*b_h + a_h*b_l),
// ~fp32 argmin fidelity at ~2.7x the exact f32-WMMA path. The codebook split is
// precomputed ONCE into fragment-ready layout so the hot loop has no conversion VALU.

#define K_CODES 1024
#define DIM     256
#define HW      1024
#define NPTS    32768
#define Q_ELEMS (32 * 256 * 1024)     // 8388608
#define LOSS_OFF Q_ELEMS
#define IDX_OFF  (Q_ELEMS + 1)
#define NWAVES  (NPTS / 16)           // 2048

// A-fragment array: [ct=64][kc=8][lane=32][slot=16] bf16  -> 512 KB each (hi, lo)
#define AFRAG_ELEMS (64 * 8 * 32 * 16)          // 262144

typedef __attribute__((ext_vector_type(16))) __bf16 v16bf;
typedef __attribute__((ext_vector_type(8)))  float  v8f;

// ---- kernel 0a: codebook -> fragment-ready bf16 hi/lo (ISA A-matrix layout) ----
__global__ void vq_prep_afrag(const float* __restrict__ cb,
                              __bf16* __restrict__ aHi,
                              __bf16* __restrict__ aLo) {
  int id = blockIdx.x * 256 + threadIdx.x;        // one element per thread
  int slot = id & 15;
  int lane = (id >> 4) & 31;
  int ckc  = id >> 9;                             // ct*8 + kc
  int kc = ckc & 7, ct = ckc >> 3;
  int m = lane & 15, half = lane >> 4;
  // A layout (16-bit 16x32): lane = 16*half + m; slot s -> vgpr j = s>>1;
  // K = (j>>2)*16 + half*8 + (j&3)*2 + (s&1)
  int j = slot >> 1;
  int k = ((j >> 2) << 4) + (half << 3) + ((j & 3) << 1) + (slot & 1);
  float f = cb[(size_t)(ct * 16 + m) * DIM + kc * 32 + k];
  __bf16 h = (__bf16)f;
  aHi[id] = h;
  aLo[id] = (__bf16)(f - (float)h);
}

// ---- kernel 0b: pre-scaled codebook norms: cn[k] = -0.5 * ||c_k||^2 ------------
__global__ void vq_cnorm_kernel(const float* __restrict__ cb,
                                float* __restrict__ cn) {
  int k = blockIdx.x * blockDim.x + threadIdx.x;
  if (k >= K_CODES) return;
  const float* row = cb + (size_t)k * DIM;
  float s = 0.f;
#pragma unroll 8
  for (int d = 0; d < DIM; ++d) { float f = row[d]; s += f * f; }
  cn[k] = -0.5f * s;
}

// ---- kernel 1: argmin + quantized scatter + loss partials ----------------------
__global__ __launch_bounds__(256) void vq_main_kernel(
    const float* __restrict__ latents, const float* __restrict__ cb,
    const v16bf* __restrict__ aHi, const v16bf* __restrict__ aLo,
    const float* __restrict__ cn, float* __restrict__ out,
    float* __restrict__ partial) {
  __shared__ __align__(32) float s_cn[K_CODES];   // holds -0.5*||c||^2
  for (int i = threadIdx.x; i < K_CODES; i += 256) s_cn[i] = cn[i];
  __syncthreads();

  const int lane = threadIdx.x & 31;
  const int wave = blockIdx.x * 8 + (threadIdx.x >> 5);
  const int nl   = lane & 15;          // point within wave's 16-point strip
  const int half = lane >> 4;
  const int n    = wave * 16 + nl;     // global point id
  const int b    = n >> 10;            // batch (HW = 1024, 16 | 1024)
  const int hw   = n & (HW - 1);
  const float* xcol = latents + (size_t)b * (DIM * HW) + hw;  // x[n][d] = xcol[d*HW]

  // ---- B fragments (x, 32x16 per chunk): bf16 hi/lo split, once per wave.
  // B layout (16-bit 32x16): lane = 16*half + n; K = (slot>>3)*16 + half*8 + (slot&7).
  v16bf bh[8], bl[8];
  float xnorm = 0.f;
#pragma unroll
  for (int kc = 0; kc < 8; ++kc) {
#pragma unroll
    for (int s = 0; s < 16; ++s) {
      int d = kc * 32 + ((s >> 3) << 4) + (half << 3) + (s & 7);
      float f = xcol[(size_t)d * HW];
      __bf16 h = (__bf16)f;
      bh[kc][s] = h;
      bl[kc][s] = (__bf16)(f - (float)h);
      xnorm += f * f;
    }
  }

  // ---- hot loop: acc init = -0.5||c||^2, so final acc = x.c - 0.5||c||^2 and
  //      d2 = -2*acc  =>  argmin(d2) == argmax(acc).
  float maxv = -3.4e38f;
  int   maxi = 0;
  for (int ct = 0; ct < 64; ++ct) {
    v8f acc;
#pragma unroll
    for (int j = 0; j < 8; ++j) acc[j] = s_cn[ct * 16 + (half << 3) + j];
    const v16bf* pH = aHi + (size_t)(ct * 8) * 32 + lane;
    const v16bf* pL = aLo + (size_t)(ct * 8) * 32 + lane;
#pragma unroll
    for (int kc = 0; kc < 8; ++kc) {
      v16bf ah = pH[kc * 32];          // 2x global_load_b128, fully coalesced
      v16bf al = pL[kc * 32];
      acc = __builtin_amdgcn_wmma_f32_16x16x32_bf16(false, ah, false, bh[kc], (short)0, acc, false, false);
      acc = __builtin_amdgcn_wmma_f32_16x16x32_bf16(false, al, false, bh[kc], (short)0, acc, false, false);
      acc = __builtin_amdgcn_wmma_f32_16x16x32_bf16(false, ah, false, bl[kc], (short)0, acc, false, false);
    }
    // D layout: lane = 16*(m>>3) + n ; vgpr j -> code = ct*16 + half*8 + j
#pragma unroll
    for (int j = 0; j < 8; ++j) {
      if (acc[j] > maxv) { maxv = acc[j]; maxi = ct * 16 + (half << 3) + j; }
    }
  }

  // cross-half merge (lane L and L+16 hold same point, disjoint code subsets)
  {
    float ov = __shfl_xor(maxv, 16, 32);
    int   oi = __shfl_xor(maxi, 16, 32);
    if (ov > maxv || (ov == maxv && oi < maxi)) { maxv = ov; maxi = oi; }
  }
  float xn = xnorm + __shfl_xor(xnorm, 16, 32);   // halves cover disjoint d-ranges

  // encoding indices (stored as float; d_out is f32)
  if (half == 0) out[IDX_OFF + n] = (float)maxi;

  // loss partial: ||q - x||^2 = d2min + ||x||^2 = -2*maxv + ||x||^2
  float contrib = (half == 0) ? (xn - 2.0f * maxv) : 0.f;
#pragma unroll
  for (int off = 8; off >= 1; off >>= 1) contrib += __shfl_xor(contrib, off, 32);
  if (lane == 0) partial[wave] = contrib;

  // quantized (straight-through value == q): out[b, d, hw] = cb[maxi, d]
  const float* qrow = cb + (size_t)maxi * DIM;
  float* oq = out + (size_t)b * (DIM * HW) + hw;
#pragma unroll 4
  for (int it = 0; it < 128; ++it) {
    int d = it * 2 + half;             // two contiguous 64B runs per store wave-wide
    oq[(size_t)d * HW] = qrow[d];
  }
}

// ---- kernel 2: deterministic loss reduction ------------------------------------
__global__ void vq_loss_reduce(const float* __restrict__ partial,
                               float* __restrict__ out) {
  __shared__ float sh[256];
  float s = 0.f;
  for (int i = threadIdx.x; i < NWAVES; i += 256) s += partial[i];
  sh[threadIdx.x] = s;
  __syncthreads();
  for (int st = 128; st >= 1; st >>= 1) {
    if (threadIdx.x < st) sh[threadIdx.x] += sh[threadIdx.x + st];
    __syncthreads();
  }
  // vq_loss = beta*mse + mse = 1.25 * sum / (N*D)
  if (threadIdx.x == 0) out[LOSS_OFF] = sh[0] * 1.25f / (float)Q_ELEMS;
}

extern "C" void kernel_launch(void* const* d_in, const int* in_sizes, int n_in,
                              void* d_out, int out_size, void* d_ws, size_t ws_size,
                              hipStream_t stream) {
  const float* latents = (const float*)d_in[0];   // [32,256,32,32] f32
  const float* cb      = (const float*)d_in[1];   // [1024,256] f32
  float* out = (float*)d_out;

  // workspace layout (bytes): [0,512K) aHi | [512K,1M) aLo | [1M,1M+4K) cn | partials
  char* ws = (char*)d_ws;
  __bf16* aHi    = (__bf16*)(ws);
  __bf16* aLo    = (__bf16*)(ws + (size_t)AFRAG_ELEMS * 2);
  float*  cn     = (float*)(ws + (size_t)AFRAG_ELEMS * 4);
  float*  partial= (float*)(ws + (size_t)AFRAG_ELEMS * 4 + K_CODES * 4);

  vq_prep_afrag<<<AFRAG_ELEMS / 256, 256, 0, stream>>>(cb, aHi, aLo);
  vq_cnorm_kernel<<<4, 256, 0, stream>>>(cb, cn);
  vq_main_kernel<<<NWAVES / 8, 256, 0, stream>>>(latents, cb,
                                                 (const v16bf*)aHi, (const v16bf*)aLo,
                                                 cn, out, partial);
  vq_loss_reduce<<<1, 256, 0, stream>>>(partial, out);
}
